// LSTMClassifier_22574348108003
// MI455X (gfx1250) — matrix operand order repaired and checked by reading, compiled, not run on hardware
//
#include <hip/hip_runtime.h>

// ---------------- problem constants ----------------
#define B_   256
#define T_   1024
#define D_   256
#define H_   512
#define L_   2
#define G4H_ 2048   // 4*H

#define NBLOCKS_ 64
#define KTOT_ 768                         // D_ + H_ per gate-column staged in LDS
#define LDS_BYTES_ (4 * 16 * KTOT_ * 2)   // 98304 bytes

typedef __bf16 bf16;
typedef __attribute__((ext_vector_type(16))) __bf16 v16bf;
typedef __attribute__((ext_vector_type(8)))  __bf16 v8bf;
typedef __attribute__((ext_vector_type(8)))  float  v8f;

// ---------------- fragment helpers ----------------
__device__ __forceinline__ v16bf make_frag(v8bf lo, v8bf hi) {
    v16bf r;
#pragma unroll
    for (int i = 0; i < 8; ++i) { r[i] = lo[i]; r[i + 8] = hi[i]; }
    return r;
}

// A fragment (16x32 bf16, MxK), row-major source.
// lane<16 -> row = lane,    K = kk+0..15
// lane>=16 -> row = lane-16, K = kk+16..31
__device__ __forceinline__ v16bf load_a(const bf16* __restrict__ row_ptr, int koff) {
    v8bf lo = *(const v8bf*)(row_ptr + koff);
    v8bf hi = *(const v8bf*)(row_ptr + koff + 8);
    return make_frag(lo, hi);
}

// B fragment (32x16, KxN) from a K-contiguous row:
// lane holds column n = lane&15, K = kk + (lane>>4)*16 + 0..15 contiguous.
__device__ __forceinline__ v16bf load_b_lds(const bf16* wrow_ptr, int koff) {
    v8bf lo = *(const v8bf*)(wrow_ptr + koff);
    v8bf hi = *(const v8bf*)(wrow_ptr + koff + 8);
    return make_frag(lo, hi);
}

__device__ __forceinline__ float sigmoidf_(float x) { return 1.0f / (1.0f + __expf(-x)); }

// ---------------- prep kernels ----------------
__global__ void cvt_f32_bf16_kernel(const float* __restrict__ src, bf16* __restrict__ dst, int n) {
    int i = blockIdx.x * blockDim.x + threadIdx.x;
    int stride = gridDim.x * blockDim.x;
    for (; i < n; i += stride) dst[i] = (bf16)src[i];
}

__global__ void init_state_kernel(bf16* __restrict__ h0, bf16* __restrict__ h1,
                                  float* __restrict__ c, unsigned* __restrict__ barrier_cnt,
                                  int n) {
    int i = blockIdx.x * blockDim.x + threadIdx.x;
    if (i < n) { h0[i] = (bf16)0.0f; h1[i] = (bf16)0.0f; c[i] = 0.0f; }
    if (i == 0) *barrier_cnt = 0u;
}

// ---------------- persistent fused LSTM ----------------
// 64 blocks x 8 waves. Block b: column tile nt = b>>1 (16 cols of H across all
// 4 gates), row half (b&1). Weight slab (96KB) staged in LDS ONCE via
// async-to-LDS DMA, then all 1024 timesteps run inside the kernel with a
// device-scope atomic grid barrier between steps. Each wave owns one 16x16
// h-tile: 4 gate accumulators -> fused sigmoid/tanh cell update in-register.
__global__ __launch_bounds__(256)
void lstm_persistent_kernel(const bf16* __restrict__ xb,   // [B, T, D] bf16
                            const bf16* __restrict__ wih,  // [4H, D]   bf16
                            const bf16* __restrict__ whh,  // [4H, H]   bf16
                            const float* __restrict__ bih, // [4H]
                            const float* __restrict__ bhh, // [4H]
                            bf16* __restrict__ h0,         // [B, H] bf16 (ping)
                            bf16* __restrict__ h1,         // [B, H] bf16 (pong)
                            float* __restrict__ c,         // [B, H] f32
                            const float* __restrict__ wout,// [L, H]
                            const float* __restrict__ bout,// [L]
                            float* __restrict__ out,       // [B, L]
                            unsigned* __restrict__ barrier_cnt)
{
    extern __shared__ bf16 smem[];   // 4*16*768 bf16 = 98304 bytes

    const int tid   = threadIdx.x;
    const int lane  = tid & 31;
    const int widx  = tid >> 5;                       // wave in block: 0..7
    const int nt    = blockIdx.x >> 1;                // 0..31 column tile of H
    const int mt    = ((blockIdx.x & 1) << 3) + widx; // 0..15 row tile of B
    const int rbase = mt << 4;
    const int cbase = nt << 4;

    // ---- stage weight slab into LDS once (async global->LDS DMA) ----
    // 98304 bytes = 6144 x 16B chunks; 256 threads x 24 chunks.
    {
        const unsigned smembase = (unsigned)(size_t)(&smem[0]);
#pragma unroll
        for (int j = 0; j < 24; ++j) {
            const int ci     = tid + j * 256;     // chunk id, coalesced
            const int row    = ci / 96;           // (g*16 + col), 0..63
            const int within = ci % 96;           // 16B chunk within 1536B row
            const int n      = ((row >> 4) << 9) + cbase + (row & 15); // weight row
            const bf16* src  = (within < 32)
                                 ? (wih + (size_t)n * D_ + within * 8)
                                 : (whh + (size_t)n * H_ + (within - 32) * 8);
            const unsigned dst = smembase + (unsigned)ci * 16u;
            asm volatile("global_load_async_to_lds_b128 %0, %1, off"
                         :: "v"(dst), "v"(src) : "memory");
        }
        asm volatile("s_wait_asynccnt 0" ::: "memory");
        __syncthreads();
    }

    const int lhalf = lane >> 4;   // 0 or 1
    const int l15   = lane & 15;

    const int arow  = rbase + l15;        // A-fragment row
    const int akoff = lhalf << 4;         // A: +0 / +16
    const int bkoff = lhalf << 4;         // B: +0 / +16

    // LDS row pointers for this lane's B column, per gate
    const bf16* w0 = smem + (size_t)(0 * 16 + l15) * KTOT_;
    const bf16* w1 = smem + (size_t)(1 * 16 + l15) * KTOT_;
    const bf16* w2 = smem + (size_t)(2 * 16 + l15) * KTOT_;
    const bf16* w3 = smem + (size_t)(3 * 16 + l15) * KTOT_;

    // per-column fused biases (loop-invariant across timesteps)
    const int bcol = cbase + l15;
    const float bg0 = bih[0 * H_ + bcol] + bhh[0 * H_ + bcol];
    const float bg1 = bih[1 * H_ + bcol] + bhh[1 * H_ + bcol];
    const float bg2 = bih[2 * H_ + bcol] + bhh[2 * H_ + bcol];
    const float bg3 = bih[3 * H_ + bcol] + bhh[3 * H_ + bcol];
    const int mofs = lhalf << 3;

    const bf16* aptr_x0 = xb + (size_t)arow * T_ * D_;  // x[arow, 0, :]
    const bf16* aptr_h  = nullptr;

    for (int t = 0; t < T_; ++t) {
        const bf16* hin = (t & 1) ? h1 : h0;
        bf16*       hou = (t & 1) ? h0 : h1;

        v8f acc0 = {}, acc1 = {}, acc2 = {}, acc3 = {};  // i, f, g, o

        // ---- phase 1: input projection, K over D; A = x[arow, t, :] ----
        {
            const bf16* aptr = aptr_x0 + (size_t)t * D_;
#pragma unroll 4
            for (int kk = 0; kk < D_; kk += 32) {
                v16bf f0 = load_b_lds(w0, kk + bkoff);
                v16bf f1 = load_b_lds(w1, kk + bkoff);
                v16bf f2 = load_b_lds(w2, kk + bkoff);
                v16bf f3 = load_b_lds(w3, kk + bkoff);
                v16bf a  = load_a(aptr, kk + akoff);
                acc0 = __builtin_amdgcn_wmma_f32_16x16x32_bf16(false, a, false, f0, (short)0, acc0, false, false);
                acc1 = __builtin_amdgcn_wmma_f32_16x16x32_bf16(false, a, false, f1, (short)0, acc1, false, false);
                acc2 = __builtin_amdgcn_wmma_f32_16x16x32_bf16(false, a, false, f2, (short)0, acc2, false, false);
                acc3 = __builtin_amdgcn_wmma_f32_16x16x32_bf16(false, a, false, f3, (short)0, acc3, false, false);
            }
        }

        // ---- phase 2: recurrent projection, K over H; A = hin[arow, :] ----
        {
            aptr_h = hin + (size_t)arow * H_;
#pragma unroll 4
            for (int kk = 0; kk < H_; kk += 32) {
                v16bf f0 = load_b_lds(w0 + D_, kk + bkoff);
                v16bf f1 = load_b_lds(w1 + D_, kk + bkoff);
                v16bf f2 = load_b_lds(w2 + D_, kk + bkoff);
                v16bf f3 = load_b_lds(w3 + D_, kk + bkoff);
                v16bf a  = load_a(aptr_h, kk + akoff);
                acc0 = __builtin_amdgcn_wmma_f32_16x16x32_bf16(false, a, false, f0, (short)0, acc0, false, false);
                acc1 = __builtin_amdgcn_wmma_f32_16x16x32_bf16(false, a, false, f1, (short)0, acc1, false, false);
                acc2 = __builtin_amdgcn_wmma_f32_16x16x32_bf16(false, a, false, f2, (short)0, acc2, false, false);
                acc3 = __builtin_amdgcn_wmma_f32_16x16x32_bf16(false, a, false, f3, (short)0, acc3, false, false);
            }
        }

        // ---- epilogue: gate activations + cell update (fused, in-register) ----
        // C/D layout: lane<16 -> N=lane, M = rbase+v ; lane>=16 -> N=lane-16, M = rbase+8+v
#pragma unroll
        for (int v = 0; v < 8; ++v) {
            const int m   = rbase + mofs + v;
            const int idx = m * H_ + bcol;
            const float gi = acc0[v] + bg0;
            const float gf = acc1[v] + bg1;
            const float gg = acc2[v] + bg2;
            const float go = acc3[v] + bg3;
            const float cn = sigmoidf_(gf) * c[idx] + sigmoidf_(gi) * tanhf(gg);
            c[idx]   = cn;
            hou[idx] = (bf16)(sigmoidf_(go) * tanhf(cn));
        }

        // ---- grid-wide barrier between timesteps ----
        __threadfence();                       // flush h writes to device scope
        __syncthreads();                       // whole block arrived
        if (tid == 0) {
            __hip_atomic_fetch_add(barrier_cnt, 1u, __ATOMIC_ACQ_REL,
                                   __HIP_MEMORY_SCOPE_AGENT);
            const unsigned target = (unsigned)NBLOCKS_ * (unsigned)(t + 1);
            while (__hip_atomic_load(barrier_cnt, __ATOMIC_ACQUIRE,
                                     __HIP_MEMORY_SCOPE_AGENT) < target) {
                __builtin_amdgcn_s_sleep(2);
            }
        }
        __syncthreads();
        __threadfence();                       // discard stale cached h before next step
    }

    // ---- classifier head: block 0 computes sigmoid(h_T @ Wout^T + bout) ----
    if (blockIdx.x == 0) {
        const bf16* hf = ((T_ - 1) & 1) ? h0 : h1;   // final hidden state
        for (int idx = tid; idx < B_ * L_; idx += 256) {
            const int b = idx >> 1;
            const int l = idx & 1;
            float s = bout[l];
            const bf16*  hp = hf + (size_t)b * H_;
            const float* wp = wout + (size_t)l * H_;
            for (int j = 0; j < H_; ++j) s += (float)hp[j] * wp[j];
            out[b * L_ + l] = 1.0f / (1.0f + __expf(-s));
        }
    }
}

// ---------------- launcher ----------------
extern "C" void kernel_launch(void* const* d_in, const int* in_sizes, int n_in,
                              void* d_out, int out_size, void* d_ws, size_t ws_size,
                              hipStream_t stream) {
    const float* x    = (const float*)d_in[0];  // [B,T,D]
    const float* Wih  = (const float*)d_in[1];  // [4H,D]
    const float* Whh  = (const float*)d_in[2];  // [4H,H]
    const float* bih  = (const float*)d_in[3];  // [4H]
    const float* bhh  = (const float*)d_in[4];  // [4H]
    const float* Wout = (const float*)d_in[5];  // [L,H]
    const float* bout = (const float*)d_in[6];  // [L]
    float* out = (float*)d_out;                 // [B,L]

    // workspace layout (bytes)
    char* ws = (char*)d_ws;
    const size_t N_X   = (size_t)B_ * T_ * D_;     // 67,108,864
    const size_t N_WIH = (size_t)G4H_ * D_;        // 524,288
    const size_t N_WHH = (size_t)G4H_ * H_;        // 1,048,576
    const size_t N_HC  = (size_t)B_ * H_;          // 131,072

    size_t off = 0;
    bf16*  xb   = (bf16*)(ws + off);  off += N_X   * sizeof(bf16);
    bf16*  wihb = (bf16*)(ws + off);  off += N_WIH * sizeof(bf16);
    bf16*  whhb = (bf16*)(ws + off);  off += N_WHH * sizeof(bf16);
    bf16*  h0   = (bf16*)(ws + off);  off += N_HC  * sizeof(bf16);
    bf16*  h1   = (bf16*)(ws + off);  off += N_HC  * sizeof(bf16);
    float* c    = (float*)(ws + off); off += N_HC  * sizeof(float);
    unsigned* barrier_cnt = (unsigned*)(ws + off); off += 256;  // padded slot
    (void)ws_size;

    // prep: f32 -> bf16 conversions + state/barrier init
    cvt_f32_bf16_kernel<<<4096, 256, 0, stream>>>(x,   xb,   (int)N_X);
    cvt_f32_bf16_kernel<<<512,  256, 0, stream>>>(Wih, wihb, (int)N_WIH);
    cvt_f32_bf16_kernel<<<1024, 256, 0, stream>>>(Whh, whhb, (int)N_WHH);
    init_state_kernel<<<(int)((N_HC + 255) / 256), 256, 0, stream>>>(h0, h1, c, barrier_cnt, (int)N_HC);

    // one persistent kernel runs the whole recurrence + classifier head
    lstm_persistent_kernel<<<NBLOCKS_, 256, LDS_BYTES_, stream>>>(
        xb, wihb, whhb, bih, bhh, h0, h1, c, Wout, bout, out, barrier_cnt);
}
